// Pointnet_FP_75282186764343
// MI455X (gfx1250) — compile-verified, matrix operand order checked by reading, not run on hardware
//
#include <hip/hip_runtime.h>
#include <stdint.h>

// ---------------------------------------------------------------------------
// Problem constants (from reference setup_inputs)
// ---------------------------------------------------------------------------
#define BS   4
#define NTGT 16384
#define NSRC 4096
#define CSRC 256
#define CTGT 128
#define CIN  384          // CSRC + CTGT (concat width)
#define CH   256          // hidden / output width
#define ROWS (BS * NTGT)  // 65536

typedef __attribute__((ext_vector_type(16))) __bf16 v16bf;
typedef __attribute__((ext_vector_type(8)))  float  v8f;

// Async global->LDS staging (CDNA5): use if the toolchain declares it.
#if defined(__has_builtin)
#if __has_builtin(__builtin_amdgcn_global_load_async_to_lds_b128) && \
    __has_builtin(__builtin_amdgcn_s_wait_asynccnt)
#define USE_ASYNC_LDS 1
#endif
#endif

#ifdef USE_ASYNC_LDS
// builtin prototype: (int4 addrspace(1)*, int4 addrspace(3)*, imm offset, imm cpol)
typedef __attribute__((ext_vector_type(4))) int vi4;
typedef __attribute__((address_space(1))) vi4 as1_vi4;
typedef __attribute__((address_space(3))) vi4 as3_vi4;
#endif

// fp32 -> bf16 with round-to-nearest-even
__device__ __forceinline__ unsigned short f2b(float f) {
  unsigned u = __builtin_bit_cast(unsigned, f);
  unsigned r = u + 0x7FFFu + ((u >> 16) & 1u);
  return (unsigned short)(r >> 16);
}

// ---------------------------------------------------------------------------
// Kernel 1: convert + pre-swizzle weights into WMMA B-fragment layout.
// B (K x 16) 16-bit layout per ISA 7.12.2: lane = h*16 + n%16 (h = (k%16)/8),
// VGPR v = (k/16)*4 + ((k%8)/2), bf16 slot = k%2.  One fragment = 1 KB,
// stored so each lane reads its 32 bytes contiguously (two b128 loads).
// Fragment order: f = kstep*16 + ntile  (ntile halves contiguous per kstep).
// ---------------------------------------------------------------------------
__global__ __launch_bounds__(256) void pack_weights(
    const float* __restrict__ w1, const float* __restrict__ w2,
    unsigned short* __restrict__ w1p, unsigned short* __restrict__ w2p) {
  int i = blockIdx.x * blockDim.x + threadIdx.x;
  const int n1 = CIN * CH;  // 98304
  const int n2 = CH * CH;   // 65536
  const float* src;
  unsigned short* dst;
  if (i < n1) {
    src = w1; dst = w1p;
  } else if (i < n1 + n2) {
    i -= n1; src = w2; dst = w2p;
  } else {
    return;
  }
  int k = i / CH, n = i % CH;
  float val = src[(size_t)k * CH + n];
  int ks = k >> 5, kk = k & 31, nt = n >> 4, nl = n & 15;
  int hi = kk >> 4, rem = kk & 15, h = rem >> 3, r2 = rem & 7;
  int v = hi * 4 + (r2 >> 1), par = r2 & 1;
  int lane = h * 16 + nl;
  int frag = ks * 16 + nt;
  dst[(size_t)frag * 512 + lane * 16 + v * 2 + par] = f2b(val);
}

// ---------------------------------------------------------------------------
// Kernel 2: 3-NN over LDS-resident source cloud + weighted gather + concat,
// emitting the bf16 activation matrix X (ROWS x CIN, row-major).
// One block = 256 targets of one batch; all 4096 source xyz live in LDS.
// Hot loop: 4 candidates per iteration via float4 LDS loads (3 ds_load_b128
// per 4 pairs), one threshold test vs. current 3rd-best on the common path.
// ---------------------------------------------------------------------------
__global__ __launch_bounds__(256) void knn_interp_concat(
    const float* __restrict__ xyz_t, const float* __restrict__ xyz_s,
    const float* __restrict__ feats_t, const float* __restrict__ feats_s,
    unsigned short* __restrict__ Xbf) {
  __shared__ __align__(16) float sx[NSRC];
  __shared__ __align__(16) float sy[NSRC];
  __shared__ __align__(16) float sz[NSRC];
  __shared__ int   s_idx[256][3];
  __shared__ float s_wgt[256][3];

  const int tid = threadIdx.x;
  const int b   = blockIdx.y;
  const int t0  = blockIdx.x * 256;

  // cooperative load of the full source cloud (48 KB) into LDS
  const float* xs = xyz_s + (size_t)b * NSRC * 3;
  for (int i = tid; i < NSRC * 3; i += 256) {
    float v = xs[i];
    int p = i / 3, c = i - p * 3;
    if (c == 0) sx[p] = v;
    else if (c == 1) sy[p] = v;
    else sz[p] = v;
  }
  __syncthreads();

  // phase 1: per-thread brute-force top-3 (squared distance; sqrt only at end)
  const int gt = t0 + tid;
  const float* tp = xyz_t + ((size_t)b * NTGT + gt) * 3;
  const float tx = tp[0], ty = tp[1], tz = tp[2];
  float b0 = 3.0e38f, b1 = 3.0e38f, b2 = 3.0e38f;
  int   i0 = 0, i1 = 0, i2 = 0;

  auto ins = [&](float d, int j) {
    if (d < b2) {
      if (d < b0)      { b2 = b1; i2 = i1; b1 = b0; i1 = i0; b0 = d; i0 = j; }
      else if (d < b1) { b2 = b1; i2 = i1; b1 = d;  i1 = j; }
      else             { b2 = d;  i2 = j; }
    }
  };

  for (int j = 0; j < NSRC; j += 4) {
    float4 X = *(const float4*)&sx[j];
    float4 Y = *(const float4*)&sy[j];
    float4 Z = *(const float4*)&sz[j];
    float dxa = tx - X.x, dya = ty - Y.x, dza = tz - Z.x;
    float dxb = tx - X.y, dyb = ty - Y.y, dzb = tz - Z.y;
    float dxc = tx - X.z, dyc = ty - Y.z, dzc = tz - Z.z;
    float dxd = tx - X.w, dyd = ty - Y.w, dzd = tz - Z.w;
    float da = dxa * dxa + dya * dya + dza * dza;
    float db = dxb * dxb + dyb * dyb + dzb * dzb;
    float dc = dxc * dxc + dyc * dyc + dzc * dzc;
    float dd = dxd * dxd + dyd * dyd + dzd * dzd;
    float m4 = fminf(fminf(da, db), fminf(dc, dd));
    if (m4 < b2) {  // rare path; insertion in index order preserves ties
      ins(da, j + 0);
      ins(db, j + 1);
      ins(dc, j + 2);
      ins(dd, j + 3);
    }
  }

  float d0 = fmaxf(sqrtf(b0), 1e-10f);
  float d1 = fmaxf(sqrtf(b1), 1e-10f);
  float d2 = fmaxf(sqrtf(b2), 1e-10f);
  float in0 = 1.0f / d0, in1 = 1.0f / d1, in2 = 1.0f / d2;
  // weights normalized; reference then divides by (sum(weight)+1e-6) == 1+1e-6
  float scale = (1.0f / (in0 + in1 + in2)) * (1.0f / (1.0f + 1e-6f));
  s_idx[tid][0] = i0; s_idx[tid][1] = i1; s_idx[tid][2] = i2;
  s_wgt[tid][0] = in0 * scale; s_wgt[tid][1] = in1 * scale; s_wgt[tid][2] = in2 * scale;
  __syncthreads();

  // phase 2: cooperative coalesced gather -> interp (256 ch) ++ feats_t (128 ch)
  const float* fs = feats_s + (size_t)b * NSRC * CSRC;
  const float* ft = feats_t + (size_t)b * NTGT * CTGT;
  for (int tl = 0; tl < 256; ++tl) {
    int j0 = s_idx[tl][0], j1 = s_idx[tl][1], j2 = s_idx[tl][2];
    float w0 = s_wgt[tl][0], w1v = s_wgt[tl][1], w2v = s_wgt[tl][2];
    size_t row = (size_t)b * NTGT + t0 + tl;
    unsigned short* xr = Xbf + row * CIN;
    float acc = w0  * fs[(size_t)j0 * CSRC + tid]
              + w1v * fs[(size_t)j1 * CSRC + tid]
              + w2v * fs[(size_t)j2 * CSRC + tid];
    xr[tid] = f2b(acc);
    if (tid < CTGT)
      xr[CSRC + tid] = f2b(ft[(size_t)(t0 + tl) * CTGT + tid]);
  }
}

// ---------------------------------------------------------------------------
// Kernels 3/4: bf16 WMMA GEMM, Y = relu(A @ B), N = 256.
// Block = 256 threads = 8 waves; wave owns one 16-row M-tile x 128 columns
// (blockIdx.y picks the N half).  All 8 waves share identical B fragments,
// so B is staged per-block into LDS, double-buffered in chunks of 2 K-steps
// (16 KB/chunk, 32 KB total) via async global->LDS copies; WMMAs then feed
// from short-latency ds_load_b128 instead of per-wave L2 round trips.
// A-fragment (16-bit 16x32, ISA 7.12.2): lane m=lane&15, h=lane>>4;
// VGPRs 0..3 = A[m][ks*32 + h*8 .. +7], VGPRs 4..7 = same +16  -> 2x b128.
// ---------------------------------------------------------------------------
template <int K, bool OUT_BF16>
__global__ __launch_bounds__(256) void gemm_wmma(
    const unsigned short* __restrict__ A,
    const unsigned short* __restrict__ Bp,
    void* __restrict__ Out) {
  constexpr int KSTEPS    = K / 32;
  constexpr int CHUNK     = 2;                    // K-steps per LDS chunk
  constexpr int NCHUNK    = KSTEPS / CHUNK;
  constexpr int SLICE_USH = 8 * 512;              // one K-step, 8 frags = 8 KB
  constexpr int CHUNK_USH = CHUNK * SLICE_USH;    // 16 KB

  __shared__ unsigned short sB[2][CHUNK_USH];

  const int tid   = threadIdx.x;
  const int wave  = tid >> 5;
  const int lane  = tid & 31;
  const int mtile = blockIdx.x * 8 + wave;
  const int nhalf = blockIdx.y;
  const int m = lane & 15, h = lane >> 4;
  const int row0 = mtile * 16;

  const unsigned short* arow = A + (size_t)(row0 + m) * K + h * 8;

  // Stage chunk c of this block's B panel into LDS buffer `buf`.
  // K-step slice: frags (ks*16 + nhalf*8) .. +7 are contiguous (8 KB);
  // 256 threads x 32 B per slice.
  auto load_chunk = [&](int c, int buf) {
#pragma unroll
    for (int kl = 0; kl < CHUNK; ++kl) {
      size_t gbase = (size_t)((c * CHUNK + kl) * 16 + nhalf * 8) * 512;
      const unsigned short* g = Bp + gbase + tid * 16;   // 32 B per thread
      unsigned short* l = &sB[buf][kl * SLICE_USH + tid * 16];
#ifdef USE_ASYNC_LDS
      __builtin_amdgcn_global_load_async_to_lds_b128((as1_vi4*)g, (as3_vi4*)l,
                                                     0, 0);
      __builtin_amdgcn_global_load_async_to_lds_b128((as1_vi4*)(g + 8),
                                                     (as3_vi4*)(l + 8), 0, 0);
#else
      uint4 t0v = *(const uint4*)g;
      uint4 t1v = *(const uint4*)(g + 8);
      *(uint4*)l = t0v;
      *(uint4*)(l + 8) = t1v;
#endif
    }
  };

  v8f acc[8];
#pragma unroll
  for (int nt = 0; nt < 8; ++nt)
    acc[nt] = (v8f){0.f, 0.f, 0.f, 0.f, 0.f, 0.f, 0.f, 0.f};

  load_chunk(0, 0);
#ifdef USE_ASYNC_LDS
  __builtin_amdgcn_s_wait_asynccnt(0);
#endif
  __syncthreads();

  for (int c = 0; c < NCHUNK; ++c) {
    const int cur = c & 1;
    if (c + 1 < NCHUNK) load_chunk(c + 1, cur ^ 1);   // prefetch next chunk

#pragma unroll
    for (int kl = 0; kl < CHUNK; ++kl) {
      const int ks = c * CHUNK + kl;
      union { uint4 q[2]; v16bf v; } a;
      a.q[0] = *(const uint4*)(arow + (size_t)ks * 32);
      a.q[1] = *(const uint4*)(arow + (size_t)ks * 32 + 16);
#pragma unroll
      for (int nt = 0; nt < 8; ++nt) {
        const unsigned short* bl =
            &sB[cur][kl * SLICE_USH + nt * 512 + lane * 16];
        v16bf bfr = *(const v16bf*)bl;
        acc[nt] = __builtin_amdgcn_wmma_f32_16x16x32_bf16(
            false, a.v, false, bfr, (short)0, acc[nt], false, false);
      }
    }

#ifdef USE_ASYNC_LDS
    __builtin_amdgcn_s_wait_asynccnt(0);
#endif
    __syncthreads();
  }

  // epilogue: ReLU + store.  D layout: VGPR v -> row (row0 + v + 8*h), col n.
#pragma unroll
  for (int nt = 0; nt < 8; ++nt) {
    int n = (nhalf * 8 + nt) * 16 + (lane & 15);
#pragma unroll
    for (int v = 0; v < 8; ++v) {
      float val = fmaxf(acc[nt][v], 0.f);
      size_t off = (size_t)(row0 + v + 8 * h) * CH + n;
      if constexpr (OUT_BF16)
        ((unsigned short*)Out)[off] = f2b(val);
      else
        ((float*)Out)[off] = val;
    }
  }
}

// ---------------------------------------------------------------------------
// Launcher.  Workspace layout (bytes):
//   [0, 50331648)              Xbf : ROWS x 384 bf16
//   [50331648, 83886080)       Y1  : ROWS x 256 bf16
//   [83886080, 84082688)       W1p : packed w1 fragments
//   [84082688, 84213760)       W2p : packed w2 fragments
// ---------------------------------------------------------------------------
extern "C" void kernel_launch(void* const* d_in, const int* in_sizes, int n_in,
                              void* d_out, int out_size, void* d_ws, size_t ws_size,
                              hipStream_t stream) {
  const float* xyz_t   = (const float*)d_in[0];
  const float* xyz_s   = (const float*)d_in[1];
  const float* feats_t = (const float*)d_in[2];
  const float* feats_s = (const float*)d_in[3];
  const float* w1      = (const float*)d_in[4];
  const float* w2      = (const float*)d_in[5];

  char* ws = (char*)d_ws;
  unsigned short* Xbf = (unsigned short*)(ws);
  unsigned short* Y1  = (unsigned short*)(ws + 50331648ull);
  unsigned short* W1p = (unsigned short*)(ws + 83886080ull);
  unsigned short* W2p = (unsigned short*)(ws + 84082688ull);

  pack_weights<<<640, 256, 0, stream>>>(w1, w2, W1p, W2p);
  knn_interp_concat<<<dim3(64, BS), 256, 0, stream>>>(xyz_t, xyz_s, feats_t,
                                                      feats_s, Xbf);
  gemm_wmma<CIN, true><<<dim3(ROWS / 16 / 8, 2), 256, 0, stream>>>(Xbf, W1p,
                                                                   (void*)Y1);
  gemm_wmma<CH, false><<<dim3(ROWS / 16 / 8, 2), 256, 0, stream>>>(Y1, W2p,
                                                                   d_out);
}
